// Pamelaformer_10934986736342
// MI455X (gfx1250) — matrix operand order repaired and checked by reading, not compiled, not run on hardware
//
#include <hip/hip_runtime.h>
#include <hip/hip_bf16.h>

// ---------------------------------------------------------------------------
// Pamelaformer for MI455X (gfx1250): bf16 WMMA everywhere, flash-style
// transposed-softmax attention, LDS-staged V tiles, fused bias+ReLU GEMMs.
// Round 2: register-blocked GEMM (4x2 tiles/wave), near-scope prefetch.
// ---------------------------------------------------------------------------

typedef __attribute__((ext_vector_type(16))) __bf16 v16bf;
typedef __attribute__((ext_vector_type(8)))  __bf16 v8bf;
typedef __attribute__((ext_vector_type(8)))  float  v8f;

#define BB   8
#define SS   1024
#define DD   768
#define HH   4
#define DH   192   // per-head value dim
#define ADIM 32    // SAM attention dim

#define GM 4       // M tiles (16 rows each) per wave in GEMM
#define GN 2       // N tiles (16 cols each) per wave in GEMM

// ---- WMMA operand loaders (layouts per CDNA5 ISA 7.12.2, wave32) ----------

// A: 16(M) x 32(K) bf16, row-major in memory with leading dim `ld`.
// lanes 0-15: row m=lane, K 0-7 (V0-3) and K 16-23 (V4-7)
// lanes 16-31: row m=lane-16, K 8-15 and K 24-31
static __device__ __forceinline__ v16bf load_a16x32(const __bf16* base, int ld) {
  const int lane = threadIdx.x & 31;
  const int m  = lane & 15;
  const int kb = (lane >> 4) << 3;
  const v8bf* p0 = (const v8bf*)(base + (size_t)m * ld + kb);
  const v8bf* p1 = (const v8bf*)(base + (size_t)m * ld + kb + 16);
  v8bf lo = *p0, hi = *p1;
  v16bf r;
#pragma unroll
  for (int i = 0; i < 8; ++i) { r[i] = lo[i]; r[8 + i] = hi[i]; }
  return r;
}

// B: 32(K) x 16(N) bf16 taken from a K-contiguous, N-major buffer ("BT"):
// lane holds column n = lane&15; lanes 0-15 K 0-7 and 16-23, lanes 16-31 K 8-15 and 24-31.
static __device__ __forceinline__ v16bf load_bt32x16(const __bf16* base, int ld) {
  const int lane = threadIdx.x & 31;
  const int n  = lane & 15;
  const int kb = (lane >> 4) << 3;
  const v8bf* p = (const v8bf*)(base + (size_t)n * ld + kb);
  v8bf lo = p[0], hi = p[2];
  v16bf r;
#pragma unroll
  for (int i = 0; i < 8; ++i) { r[i] = lo[i]; r[8 + i] = hi[i]; }
  return r;
}

static __device__ __forceinline__ v8f wmma_bf16(v16bf a, v16bf b, v8f c) {
  return __builtin_amdgcn_wmma_f32_16x16x32_bf16(false, a, false, b,
                                                 (short)0, c, false, false);
}

// ---- small prep kernels ----------------------------------------------------

__global__ void cvt_f32_bf16_kernel(const float* __restrict__ s,
                                    __bf16* __restrict__ d, int n) {
  int i = blockIdx.x * blockDim.x + threadIdx.x;
  if (i < n) d[i] = (__bf16)s[i];
}

// src [R,C] f32 -> dst [C,R] bf16  (weights -> K-contiguous per output col)
__global__ void transpose_cvt_kernel(const float* __restrict__ src,
                                     __bf16* __restrict__ dst, int R, int C) {
  int c = blockIdx.x * 32 + (threadIdx.x & 31);
  int r = blockIdx.y * 8 + (threadIdx.x >> 5);
  if (r < R && c < C) dst[(size_t)c * R + r] = (__bf16)src[(size_t)r * C + c];
}

// ---- register-blocked bf16 WMMA GEMM ---------------------------------------
// C = act(A[MxK] @ BT[NxK]^T + bias). block = 128 threads = 4 waves along M;
// each wave owns a 64x32 output patch (4x2 WMMA tiles, 8 f32 accumulators).
__global__ void gemm_bf16_kernel(const __bf16* __restrict__ A,
                                 const __bf16* __restrict__ BT,
                                 const float* __restrict__ bias,
                                 __bf16* __restrict__ Cb, float* __restrict__ Cf,
                                 int K, int lda, int ldb, int ldc, int relu) {
  const int wave = threadIdx.x >> 5;
  const int lane = threadIdx.x & 31;
  const int i0 = (blockIdx.y * 4 + wave) * (16 * GM);
  const int n0 = blockIdx.x * (16 * GN);

  v8f z = {};
  v8f acc[GM][GN];
#pragma unroll
  for (int mi = 0; mi < GM; ++mi)
#pragma unroll
    for (int ni = 0; ni < GN; ++ni) acc[mi][ni] = z;

  const __bf16* ap = A  + (size_t)i0 * lda;
  const __bf16* bp = BT + (size_t)n0 * ldb;
  for (int k = 0; k < K; k += 32) {
    __builtin_prefetch(ap + k + 512, 0, 3);  // near-scope global_prefetch_b8
    __builtin_prefetch(bp + k + 512, 0, 3);
    v16bf a[GM], b[GN];
#pragma unroll
    for (int mi = 0; mi < GM; ++mi)
      a[mi] = load_a16x32(ap + (size_t)(mi * 16) * lda + k, lda);
#pragma unroll
    for (int ni = 0; ni < GN; ++ni)
      b[ni] = load_bt32x16(bp + (size_t)(ni * 16) * ldb + k, ldb);
#pragma unroll
    for (int mi = 0; mi < GM; ++mi)
#pragma unroll
      for (int ni = 0; ni < GN; ++ni)
        acc[mi][ni] = wmma_bf16(a[mi], b[ni], acc[mi][ni]);
  }

  const int n   = lane & 15;
  const int mb8 = (lane >> 4) << 3;
#pragma unroll
  for (int ni = 0; ni < GN; ++ni) {
    const float bs = bias[n0 + ni * 16 + n];
#pragma unroll
    for (int mi = 0; mi < GM; ++mi) {
#pragma unroll
      for (int r = 0; r < 8; ++r) {
        float v = acc[mi][ni][r] + bs;
        if (relu) v = fmaxf(v, 0.0f);
        size_t off = (size_t)(i0 + mi * 16 + mb8 + r) * ldc + (n0 + ni * 16 + n);
        if (Cf) Cf[off] = v;
        else    Cb[off] = (__bf16)v;
      }
    }
  }
}

// ---- SAM attention (dqk = 8, scores on VALU, PV on WMMA) -------------------
// Transposed-softmax attention: rows = keys, softmax streamed over queries.
__global__ void sam_attn_kernel(const __bf16* __restrict__ qb,
                                const __bf16* __restrict__ kbuf,
                                const __bf16* __restrict__ xb,
                                const int* __restrict__ mask,
                                __bf16* __restrict__ outp, int ldo) {
  __shared__ __bf16 vstT[DH * 32];          // V chunk transposed [c][r]
  __shared__ float  s_tile[4][16 * 32];
  __shared__ float  kblk[4][16 * 8];
  __shared__ float  rmax[4][16], rsum[4][16], rscale[4][16];
  __shared__ float  mrow[4][16];
  __shared__ float  mcol[4][32];

  const int wave = threadIdx.x >> 5;
  const int lane = threadIdx.x & 31;
  const int b = blockIdx.z, h = blockIdx.y;
  const int i0 = (blockIdx.x * 4 + wave) * 16;

  const __bf16* qh = qb   + (size_t)b * SS * ADIM + h * 8;
  const __bf16* kh = kbuf + (size_t)b * SS * ADIM + h * 8;
  const __bf16* vh = xb   + (size_t)b * SS * DD + h * DH;
  const int* mb = mask + b * SS;

  if (lane < 16) {
    rmax[wave][lane] = -3.0e38f;
    rsum[wave][lane] = 0.0f;
    mrow[wave][lane] = (float)mb[i0 + lane];
  }
  for (int t = lane; t < 16 * 8; t += 32)
    kblk[wave][t] = (float)kh[(size_t)(i0 + (t >> 3)) * ADIM + (t & 7)];

  v8f z = {};
  v8f acc[12];
#pragma unroll
  for (int t = 0; t < 12; ++t) acc[t] = z;

  const float scale = 0.59460355750136051f;  // 8^-0.25
  const int m     = lane & 15;
  const int jhalf = lane >> 4;
  const int mb8   = jhalf << 3;
  __syncthreads();

  for (int j0 = 0; j0 < SS; j0 += 32) {
    mcol[wave][lane] = (float)mb[j0 + lane];
    // stage V chunk (transposed) for all 4 waves
    for (int t = threadIdx.x; t < 32 * DH; t += 128) {
      int r = t / DH, c = t - r * DH;
      vstT[c * 32 + r] = vh[(size_t)(j0 + r) * DD + c];
    }
    // raw scores s[m][j] = k_m . q_j  (dot-8 per element)
    float kr[8];
#pragma unroll
    for (int e = 0; e < 8; ++e) kr[e] = kblk[wave][m * 8 + e];
    for (int c = 0; c < 16; ++c) {
      int j = jhalf * 16 + c;
      const v8bf qv = *(const v8bf*)(qh + (size_t)(j0 + j) * ADIM);
      float d = 0.0f;
#pragma unroll
      for (int e = 0; e < 8; ++e) d += kr[e] * (float)qv[e];
      s_tile[wave][m * 32 + j] = d;
    }
    __syncthreads();
    // online softmax over query axis (lanes 0..15 own one row each)
    if (lane < 16) {
      const float mi = mrow[wave][lane];
      const float oldmax = rmax[wave][lane];
      float mx = oldmax;
      for (int j = 0; j < 32; ++j) {
        float raw = s_tile[wave][lane * 32 + j];
        float val = (mi != 0.0f && mcol[wave][j] != 0.0f) ? raw * scale : -1.0e7f;
        mx = fmaxf(mx, val);
      }
      float corr = __expf(oldmax - mx);
      float s = rsum[wave][lane] * corr;
      for (int j = 0; j < 32; ++j) {
        float raw = s_tile[wave][lane * 32 + j];
        float val = (mi != 0.0f && mcol[wave][j] != 0.0f) ? raw * scale : -1.0e7f;
        float p = __expf(val - mx);
        s += p;
        s_tile[wave][lane * 32 + j] = p;
      }
      rmax[wave][lane] = mx; rsum[wave][lane] = s; rscale[wave][lane] = corr;
    }
    __syncthreads();
    // rescale accumulators by per-row correction
    float cr[8];
#pragma unroll
    for (int r = 0; r < 8; ++r) cr[r] = rscale[wave][mb8 + r];
#pragma unroll
    for (int t = 0; t < 12; ++t)
#pragma unroll
      for (int r = 0; r < 8; ++r) acc[t][r] *= cr[r];
    // P tile in A layout
    v16bf pa;
#pragma unroll
    for (int i = 0; i < 8; ++i) {
      pa[i]     = (__bf16)s_tile[wave][m * 32 + mb8 + i];
      pa[8 + i] = (__bf16)s_tile[wave][m * 32 + mb8 + 16 + i];
    }
    // acc += P[16x32] @ V[32x192]
#pragma unroll
    for (int t = 0; t < 12; ++t) {
      v16bf bt = load_bt32x16(vstT + (size_t)(t * 16) * 32, 32);
      acc[t] = wmma_bf16(pa, bt, acc[t]);
    }
    __syncthreads();
  }
  float inv[8];
#pragma unroll
  for (int r = 0; r < 8; ++r) inv[r] = 1.0f / rsum[wave][mb8 + r];
  const int n = lane & 15;
  __bf16* ob = outp + (size_t)b * SS * ldo + (size_t)h * DH;
#pragma unroll
  for (int t = 0; t < 12; ++t)
#pragma unroll
    for (int r = 0; r < 8; ++r)
      ob[(size_t)(i0 + mb8 + r) * ldo + t * 16 + n] = (__bf16)(acc[t][r] * inv[r]);
}

// ---- PAM attention (dqk = 192, scores AND PV on WMMA) ----------------------
__global__ void pam_attn_kernel(const __bf16* __restrict__ pb,
                                const __bf16* __restrict__ xb,
                                const int* __restrict__ mask,
                                __bf16* __restrict__ outp, int ldo) {
  __shared__ __bf16 vstT[DH * 32];
  __shared__ float  s_tile[4][16 * 32];
  __shared__ float  rmax[4][16], rsum[4][16], rscale[4][16];
  __shared__ float  mrow[4][16];
  __shared__ float  mcol[4][32];

  const int wave = threadIdx.x >> 5;
  const int lane = threadIdx.x & 31;
  const int b = blockIdx.z, h = blockIdx.y;
  const int i0 = (blockIdx.x * 4 + wave) * 16;

  const __bf16* ph = pb + (size_t)b * SS * DD + h * DH;  // [S, DH] stride DD
  const __bf16* vh = xb + (size_t)b * SS * DD + h * DH;
  const int* mb = mask + b * SS;

  if (lane < 16) {
    rmax[wave][lane] = -3.0e38f;
    rsum[wave][lane] = 0.0f;
    mrow[wave][lane] = (float)mb[i0 + lane];
  }
  // hoist A tiles: pos rows i0..i0+15 over K = 192 (6 chunks of 32)
  v16bf ap[6];
#pragma unroll
  for (int kc = 0; kc < 6; ++kc)
    ap[kc] = load_a16x32(ph + (size_t)i0 * DD + kc * 32, DD);

  v8f z = {};
  v8f acc[12];
#pragma unroll
  for (int t = 0; t < 12; ++t) acc[t] = z;

  const float scale = 0.26864244187569337f;  // 192^-0.25
  const int m     = lane & 15;
  const int jhalf = lane >> 4;
  const int mb8   = jhalf << 3;
  const int n     = lane & 15;
  __syncthreads();

  for (int j0 = 0; j0 < SS; j0 += 32) {
    mcol[wave][lane] = (float)mb[j0 + lane];
    for (int t = threadIdx.x; t < 32 * DH; t += 128) {
      int r = t / DH, c = t - r * DH;
      vstT[c * 32 + r] = vh[(size_t)(j0 + r) * DD + c];
    }
    // scores[16 x 32] = Pos_i @ Pos_j^T  via WMMA (B columns = pos rows -> contiguous)
    v8f sacc[2] = { z, z };
#pragma unroll
    for (int kc = 0; kc < 6; ++kc) {
#pragma unroll
      for (int t2 = 0; t2 < 2; ++t2) {
        v16bf bp = load_bt32x16(ph + (size_t)(j0 + t2 * 16) * DD + kc * 32, DD);
        sacc[t2] = wmma_bf16(ap[kc], bp, sacc[t2]);
      }
    }
#pragma unroll
    for (int t2 = 0; t2 < 2; ++t2)
#pragma unroll
      for (int r = 0; r < 8; ++r)
        s_tile[wave][(mb8 + r) * 32 + t2 * 16 + n] = sacc[t2][r];
    __syncthreads();
    if (lane < 16) {
      const float mi = mrow[wave][lane];
      const float oldmax = rmax[wave][lane];
      float mx = oldmax;
      for (int j = 0; j < 32; ++j) {
        float raw = s_tile[wave][lane * 32 + j];
        float val = (mi != 0.0f && mcol[wave][j] != 0.0f) ? raw * scale : -1.0e7f;
        mx = fmaxf(mx, val);
      }
      float corr = __expf(oldmax - mx);
      float s = rsum[wave][lane] * corr;
      for (int j = 0; j < 32; ++j) {
        float raw = s_tile[wave][lane * 32 + j];
        float val = (mi != 0.0f && mcol[wave][j] != 0.0f) ? raw * scale : -1.0e7f;
        float p = __expf(val - mx);
        s += p;
        s_tile[wave][lane * 32 + j] = p;
      }
      rmax[wave][lane] = mx; rsum[wave][lane] = s; rscale[wave][lane] = corr;
    }
    __syncthreads();
    float cr[8];
#pragma unroll
    for (int r = 0; r < 8; ++r) cr[r] = rscale[wave][mb8 + r];
#pragma unroll
    for (int t = 0; t < 12; ++t)
#pragma unroll
      for (int r = 0; r < 8; ++r) acc[t][r] *= cr[r];
    v16bf pa;
#pragma unroll
    for (int i = 0; i < 8; ++i) {
      pa[i]     = (__bf16)s_tile[wave][m * 32 + mb8 + i];
      pa[8 + i] = (__bf16)s_tile[wave][m * 32 + mb8 + 16 + i];
    }
#pragma unroll
    for (int t = 0; t < 12; ++t) {
      v16bf bt = load_bt32x16(vstT + (size_t)(t * 16) * 32, 32);
      acc[t] = wmma_bf16(pa, bt, acc[t]);
    }
    __syncthreads();
  }
  float inv[8];
#pragma unroll
  for (int r = 0; r < 8; ++r) inv[r] = 1.0f / rsum[wave][mb8 + r];
  __bf16* ob = outp + (size_t)b * SS * ldo + (size_t)h * DH;
#pragma unroll
  for (int t = 0; t < 12; ++t)
#pragma unroll
    for (int r = 0; r < 8; ++r)
      ob[(size_t)(i0 + mb8 + r) * ldo + t * 16 + n] = (__bf16)(acc[t][r] * inv[r]);
}

// ---------------------------------------------------------------------------

extern "C" void kernel_launch(void* const* d_in, const int* in_sizes, int n_in,
                              void* d_out, int out_size, void* d_ws, size_t ws_size,
                              hipStream_t stream) {
  (void)in_sizes; (void)n_in; (void)out_size; (void)ws_size;
  const float* x    = (const float*)d_in[0];
  const int*   mask = (const int*)  d_in[1];
  const float* pos  = (const float*)d_in[2];
  const float* Wq   = (const float*)d_in[3];
  const float* bq   = (const float*)d_in[4];
  const float* Wk   = (const float*)d_in[5];
  const float* bk   = (const float*)d_in[6];
  const float* Wpam = (const float*)d_in[7];
  const float* bpam = (const float*)d_in[8];
  const float* Wm   = (const float*)d_in[9];
  const float* bm   = (const float*)d_in[10];
  float* out = (float*)d_out;

  char* ws = (char*)d_ws;
  size_t off = 0;
  auto take = [&](size_t bytes) {
    char* p = ws + off;
    off = (off + bytes + 255) & ~(size_t)255;
    return p;
  };
  __bf16* xb    = (__bf16*)take((size_t)BB * SS * DD * 2);
  __bf16* posb  = (__bf16*)take((size_t)BB * SS * DD * 2);
  __bf16* qb    = (__bf16*)take((size_t)BB * SS * ADIM * 2);
  __bf16* kb    = (__bf16*)take((size_t)BB * SS * ADIM * 2);
  __bf16* WqT   = (__bf16*)take((size_t)ADIM * DD * 2);
  __bf16* WkT   = (__bf16*)take((size_t)ADIM * DD * 2);
  __bf16* WpamT = (__bf16*)take((size_t)DD * DD * 2);
  __bf16* WmT   = (__bf16*)take((size_t)DD * 2 * DD * 2);
  __bf16* pamA  = (__bf16*)take((size_t)BB * SS * DD * 2);
  __bf16* cat   = (__bf16*)take((size_t)BB * SS * 2 * DD * 2);

  const int NTOT = BB * SS * DD;
  cvt_f32_bf16_kernel<<<(NTOT + 255) / 256, 256, 0, stream>>>(x, xb, NTOT);
  cvt_f32_bf16_kernel<<<(NTOT + 255) / 256, 256, 0, stream>>>(pos, posb, NTOT);
  transpose_cvt_kernel<<<dim3((ADIM + 31) / 32, (DD + 7) / 8), 256, 0, stream>>>(Wq, WqT, DD, ADIM);
  transpose_cvt_kernel<<<dim3((ADIM + 31) / 32, (DD + 7) / 8), 256, 0, stream>>>(Wk, WkT, DD, ADIM);
  transpose_cvt_kernel<<<dim3((DD + 31) / 32, (DD + 7) / 8), 256, 0, stream>>>(Wpam, WpamT, DD, DD);
  transpose_cvt_kernel<<<dim3((DD + 31) / 32, (2 * DD + 7) / 8), 256, 0, stream>>>(Wm, WmT, 2 * DD, DD);

  const int M = BB * SS;
  const int MROWS = 64 * GM;  // rows per block (4 waves * 16*GM)
  // q = relu(x@Wq+bq), k = relu(x@Wk+bk)  -> bf16 [M, 32]
  gemm_bf16_kernel<<<dim3(ADIM / (16 * GN), M / MROWS), 128, 0, stream>>>(
      xb, WqT, bq, qb, nullptr, DD, DD, DD, ADIM, 1);
  gemm_bf16_kernel<<<dim3(ADIM / (16 * GN), M / MROWS), 128, 0, stream>>>(
      xb, WkT, bk, kb, nullptr, DD, DD, DD, ADIM, 1);

  // SAM attention -> concat[:, 0:768)
  sam_attn_kernel<<<dim3(SS / 64, HH, BB), 128, 0, stream>>>(
      qb, kb, xb, mask, cat, 2 * DD);
  // PAM attention -> pamA
  pam_attn_kernel<<<dim3(SS / 64, HH, BB), 128, 0, stream>>>(
      posb, xb, mask, pamA, DD);
  // pos_out = relu(pamA @ Wpam + bpam) -> concat[:, 768:1536)
  gemm_bf16_kernel<<<dim3(DD / (16 * GN), M / MROWS), 128, 0, stream>>>(
      pamA, WpamT, bpam, cat + DD, nullptr, DD, DD, DD, 2 * DD, 1);
  // c = relu(concat @ Wm + bm) -> f32 out
  gemm_bf16_kernel<<<dim3(DD / (16 * GN), M / MROWS), 128, 0, stream>>>(
      cat, WmT, bm, nullptr, out, 2 * DD, 2 * DD, 2 * DD, DD, 1);
}
